// ProtoCell_30073361006854
// MI455X (gfx1250) — compile-verified, matrix operand order checked by reading, not run on hardware
//
#include <hip/hip_runtime.h>
#include <hip/hip_bf16.h>

// ---------------------------------------------------------------------------
// Types for CDNA5 WMMA
// ---------------------------------------------------------------------------
typedef __attribute__((ext_vector_type(16))) __bf16 v16bf;
typedef __attribute__((ext_vector_type(8)))  float  v8f;
typedef __attribute__((ext_vector_type(4)))  float  v4f;

// two floats -> packed bf16 pair (single v_cvt_pk_bf16_f32 when available)
__device__ __forceinline__ unsigned int cvt2(float a, float b) {
#if __has_builtin(__builtin_amdgcn_cvt_pk_bf16_f32)
    auto r = __builtin_amdgcn_cvt_pk_bf16_f32(a, b);
    return __builtin_bit_cast(unsigned int, r);
#else
    union { unsigned int u; __bf16 e[2]; } q;
    q.e[0] = (__bf16)a; q.e[1] = (__bf16)b;
    return q.u;
#endif
}

// 8 floats -> 8 packed bf16
__device__ __forceinline__ uint4 pack8(v4f v0, v4f v1) {
    uint4 p;
    p.x = cvt2(v0.x, v0.y); p.y = cvt2(v0.z, v0.w);
    p.z = cvt2(v1.x, v1.y); p.w = cvt2(v1.z, v1.w);
    return p;
}

__device__ __forceinline__ float leaky(float v) {
    return v > 0.0f ? v : 0.01f * v;
}

// Model dims
#define NCELLS 50000
#define INDIM  2000
#define HDIM   128
#define ZDIM   64
#define NPROTO 32
#define NCLS   8
#define NB     64

// ---------------------------------------------------------------------------
// Kernel A: Ybf[N,256] = bf16( leakyrelu( X[N,2000] @ [enc_i_w ; imp_i_w]^T + b ) )
// 128x256 tile/block, K-tile 32, double-buffered LDS, branch-free main loop.
// 512 threads = 16 waves; wave w owns 2 M-subtiles x 4 N-subtiles.
// x loads are non-temporal (single-use 400MB stream; keep L2 for W and Y).
// ---------------------------------------------------------------------------
#define MT 128
#define NT 256
#define KT 32
#define NKFULL 62           // tiles 0..61 fully valid; tile 62 has 16 valid k

__global__ __launch_bounds__(512)
void fused_layer1(const float* __restrict__ x,
                  const float* __restrict__ w_enc, const float* __restrict__ b_enc,
                  const float* __restrict__ w_imp, const float* __restrict__ b_imp,
                  __bf16* __restrict__ Ybf)
{
    __shared__ __attribute__((aligned(32))) __bf16 As[2][MT][KT];
    __shared__ __attribute__((aligned(32))) __bf16 Bs[2][NT][KT];

    const int tid  = threadIdx.x;
    const int wv   = tid >> 5;
    const int lane = tid & 31;
    const int m0   = blockIdx.x * MT;

    // per-thread staging coordinates (constant across tiles)
    const int xrow = tid >> 2;
    const int xqk  = (tid & 3) * 8;
    const int xgrow = min(m0 + xrow, NCELLS - 1);          // clamp: branch-free
    const float* xbase = x + (size_t)xgrow * INDIM + xqk;

    const int wh  = tid >> 1;
    const int wkq = (tid & 1) * 16;
    const float* wrow = (wh < HDIM) ? (w_enc + (size_t)wh * INDIM)
                                    : (w_imp + (size_t)(wh - HDIM) * INDIM);

    auto load_full = [&](int kt, int buf) {
        const int k0 = kt * KT;
        {   // X tile: 8 floats (NT stream loads) -> one b128 LDS store
            v4f v0 = __builtin_nontemporal_load((const v4f*)(xbase + k0));
            v4f v1 = __builtin_nontemporal_load((const v4f*)(xbase + k0 + 4));
            *(uint4*)&As[buf][xrow][xqk] = pack8(v0, v1);
        }
        {   // W tile: 16 floats (cache-resident) -> two b128 LDS stores
            const float* src = wrow + k0 + wkq;
            #pragma unroll
            for (int h = 0; h < 2; ++h) {
                v4f v0 = *(const v4f*)(src + h * 8);
                v4f v1 = *(const v4f*)(src + h * 8 + 4);
                *(uint4*)&Bs[buf][wh][wkq + h * 8] = pack8(v0, v1);
            }
        }
    };

    auto load_partial = [&](int buf) {                      // kt = 62, k0 = 1984
        const int k0 = NKFULL * KT;
        {
            uint4 p = make_uint4(0u, 0u, 0u, 0u);
            if (xqk < 16) {                                 // k 1984..1999 valid
                v4f v0 = __builtin_nontemporal_load((const v4f*)(xbase + k0));
                v4f v1 = __builtin_nontemporal_load((const v4f*)(xbase + k0 + 4));
                p = pack8(v0, v1);
            }
            *(uint4*)&As[buf][xrow][xqk] = p;
        }
        {
            uint4 p0 = make_uint4(0u, 0u, 0u, 0u);
            uint4 p1 = make_uint4(0u, 0u, 0u, 0u);
            if (wkq == 0) {
                const float* src = wrow + k0;
                p0 = pack8(*(const v4f*)(src),     *(const v4f*)(src + 4));
                p1 = pack8(*(const v4f*)(src + 8), *(const v4f*)(src + 12));
            }
            *(uint4*)&Bs[buf][wh][wkq]     = p0;
            *(uint4*)&Bs[buf][wh][wkq + 8] = p1;
        }
    };

    v8f acc[4][2];
    #pragma unroll
    for (int ni = 0; ni < 4; ++ni) {
        acc[ni][0] = {};
        acc[ni][1] = {};
    }

    const int mrow  = wv >> 2;          // 0..3 -> M element base mrow*32
    const int ncol  = wv & 3;           // 0..3 -> N element base ncol*64
    const int ma    = mrow * 32;
    const int nbase = ncol * 64;
    const int lrow  = lane & 15;
    const int lhalf = lane >> 4;

    auto compute = [&](int buf) {
        v16bf a0 = *(const v16bf*)&As[buf][ma + lrow][lhalf * 16];
        v16bf a1 = *(const v16bf*)&As[buf][ma + 16 + lrow][lhalf * 16];
        #pragma unroll
        for (int ni = 0; ni < 4; ++ni) {
            v16bf b = *(const v16bf*)&Bs[buf][nbase + ni * 16 + lrow][lhalf * 16];
            acc[ni][0] = __builtin_amdgcn_wmma_f32_16x16x32_bf16(
                false, a0, false, b, (short)0, acc[ni][0], false, false);
            acc[ni][1] = __builtin_amdgcn_wmma_f32_16x16x32_bf16(
                false, a1, false, b, (short)0, acc[ni][1], false, false);
        }
    };

    load_full(0, 0);
    for (int kt = 0; kt < NKFULL; ++kt) {
        __syncthreads();
        if (kt + 1 < NKFULL) load_full(kt + 1, (kt + 1) & 1);
        else                 load_partial(NKFULL & 1);      // tile 62 -> buf 0
        compute(kt & 1);
    }
    __syncthreads();
    compute(NKFULL & 1);

    // epilogue: bias + leaky-ReLU, store bf16
    auto store_out = [&](bool guard) {
        #pragma unroll
        for (int ni = 0; ni < 4; ++ni) {
            int col   = nbase + ni * 16 + lrow;
            float bs  = (col < HDIM) ? b_enc[col] : b_imp[col - HDIM];
            #pragma unroll
            for (int m2 = 0; m2 < 2; ++m2) {
                #pragma unroll
                for (int i = 0; i < 8; ++i) {
                    int row = m0 + ma + m2 * 16 + i + lhalf * 8;
                    if (!guard || row < NCELLS)
                        Ybf[(size_t)row * NT + col] = (__bf16)leaky(acc[ni][m2][i] + bs);
                }
            }
        }
    };
    if (m0 + MT <= NCELLS) store_out(false);   // 390 full blocks: unguarded
    else                   store_out(true);    // last block only
}

// ---------------------------------------------------------------------------
// Kernel B: per-wave fused tail. One wave = one 16-row group.
// A-frags straight from bf16 Y in global (hoisted per layer); B-frags from
// pre-converted bf16 weights. Only hmid round-trips LDS (layout shuffle).
// ---------------------------------------------------------------------------
#define WPB 4   // waves per block (128 threads)

__global__ __launch_bounds__(128)
void fused_tail(const __bf16* __restrict__ Ybf, const int* __restrict__ seg,
                const __bf16* __restrict__ w_eh, const float* __restrict__ b_eh,
                const __bf16* __restrict__ w_ez, const float* __restrict__ b_ez,
                const __bf16* __restrict__ w_ih, const float* __restrict__ b_ih,
                const __bf16* __restrict__ w_ip, const float* __restrict__ b_ip,
                const float* __restrict__ proto,
                float* __restrict__ ssum, float* __restrict__ scnt, int ngroups)
{
    __shared__ __attribute__((aligned(32))) __bf16 hmid[WPB][16][HDIM];
    __shared__ float zbuf  [WPB][16][ZDIM];
    __shared__ float simbuf[WPB][16][NPROTO];
    __shared__ float clog  [WPB][16][NCLS];

    const int lane = threadIdx.x & 31;
    const int wv   = threadIdx.x >> 5;
    const int g    = blockIdx.x * WPB + wv;
    if (g >= ngroups) return;          // wave-uniform: EXEC stays all-1s below
    const int r0    = g * 16;
    const int lrow  = lane & 15;
    const int lhalf = lane >> 4;

    // B-fragment straight from global bf16 weights (row-major [out,128])
    auto bfragW = [&](const __bf16* __restrict__ W, int n, int kk) -> v16bf {
        return *(const v16bf*)(W + (size_t)n * HDIM + kk * 32 + lhalf * 16);
    };
    // hoist the 4 A-fragments of this wave's 16x128 input slab (from global Y)
    auto loadAY = [&](v16bf* a, int cbase) {
        const __bf16* base = Ybf + (size_t)(r0 + lrow) * NT + cbase + lhalf * 16;
        #pragma unroll
        for (int kk = 0; kk < 4; ++kk) a[kk] = *(const v16bf*)(base + kk * 32);
    };
    // hoist the 4 A-fragments of the LDS inter-layer buffer
    auto loadAM = [&](v16bf* a) {
        #pragma unroll
        for (int kk = 0; kk < 4; ++kk)
            a[kk] = *(const v16bf*)&hmid[wv][lrow][kk * 32 + lhalf * 16];
    };

    // hmid = leakyrelu(Y[:, cbase:cbase+128] @ W^T + bias)
    auto layer128 = [&](int cbase, const __bf16* __restrict__ W,
                        const float* __restrict__ bias) {
        v16bf a[4];
        loadAY(a, cbase);
        #pragma unroll
        for (int ni = 0; ni < 8; ++ni) {
            v8f acc = {};
            #pragma unroll
            for (int kk = 0; kk < 4; ++kk)
                acc = __builtin_amdgcn_wmma_f32_16x16x32_bf16(
                    false, a[kk], false, bfragW(W, ni * 16 + lrow, kk),
                    (short)0, acc, false, false);
            int col  = ni * 16 + lrow;
            float bs = bias[col];
            #pragma unroll
            for (int i = 0; i < 8; ++i)
                hmid[wv][i + lhalf * 8][col] = (__bf16)leaky(acc[i] + bs);
        }
    };

    // ---------------- encoder path ----------------
    layer128(0, w_eh, b_eh);

    // z = leakyrelu(hmid @ enc_z_w^T + b): 64 outputs, f32 in LDS
    {
        v16bf a[4];
        loadAM(a);
        #pragma unroll
        for (int ni = 0; ni < 4; ++ni) {
            v8f acc = {};
            #pragma unroll
            for (int kk = 0; kk < 4; ++kk)
                acc = __builtin_amdgcn_wmma_f32_16x16x32_bf16(
                    false, a[kk], false, bfragW(w_ez, ni * 16 + lrow, kk),
                    (short)0, acc, false, false);
            int col  = ni * 16 + lrow;
            float bs = b_ez[col];
            #pragma unroll
            for (int i = 0; i < 8; ++i)
                zbuf[wv][i + lhalf * 8][col] = leaky(acc[i] + bs);
        }
    }

    // sim[row][p] = 1/(||z - proto_p||^2 + 0.5); lane == p
    {
        float pr[ZDIM];
        #pragma unroll
        for (int d = 0; d < ZDIM; ++d) pr[d] = proto[lane * ZDIM + d];
        for (int row = 0; row < 16; ++row) {
            float d2 = 0.0f;
            #pragma unroll
            for (int d = 0; d < ZDIM; d += 4) {
                v4f zv = *(const v4f*)&zbuf[wv][row][d];
                float df0 = zv.x - pr[d];     d2 = fmaf(df0, df0, d2);
                float df1 = zv.y - pr[d + 1]; d2 = fmaf(df1, df1, d2);
                float df2 = zv.z - pr[d + 2]; d2 = fmaf(df2, df2, d2);
                float df3 = zv.w - pr[d + 3]; d2 = fmaf(df3, df3, d2);
            }
            simbuf[wv][row][lane] = 1.0f / (d2 + 0.5f);
        }
    }

    for (int idx = lane; idx < 16 * NCLS; idx += 32)
        clog[wv][idx >> 3][idx & 7] = 0.0f;

    // ---------------- importance path ----------------
    layer128(HDIM, w_ih, b_ih);

    // imp = sigmoid(hmid @ imp_p_w^T + b) -> c_logits accumulation (LDS atomics)
    {
        v16bf a[4];
        loadAM(a);
        #pragma unroll
        for (int ni = 0; ni < 16; ++ni) {
            v8f acc = {};
            #pragma unroll
            for (int kk = 0; kk < 4; ++kk)
                acc = __builtin_amdgcn_wmma_f32_16x16x32_bf16(
                    false, a[kk], false, bfragW(w_ip, ni * 16 + lrow, kk),
                    (short)0, acc, false, false);
            int col  = ni * 16 + lrow;
            float bs = b_ip[col];
            int p = col >> 3, c = col & 7;
            #pragma unroll
            for (int i = 0; i < 8; ++i) {
                int row = i + lhalf * 8;
                float v = acc[i] + bs;
                v = 1.0f / (1.0f + __expf(-v));             // sigmoid
                atomicAdd(&clog[wv][row][c], simbuf[wv][row][p] * v);
            }
        }
    }

    // segment accumulation in global memory
    for (int idx = lane; idx < 16 * NCLS; idx += 32) {
        int row = idx >> 3, c = idx & 7;
        int s = seg[r0 + row];
        atomicAdd(&ssum[s * NCLS + c], clog[wv][row][c]);
    }
    if (lane < 16) {
        int s = seg[r0 + lane];
        atomicAdd(&scnt[s], 1.0f);
    }
}

// ---------------------------------------------------------------------------
// Kernel C: segment means, log-softmax, mean cross-entropy
// d_out layout: [loss, logits(64x8)]
// ---------------------------------------------------------------------------
__global__ void finalize_kernel(const float* __restrict__ ssum,
                                const float* __restrict__ scnt,
                                const int* __restrict__ y,
                                float* __restrict__ out)
{
    __shared__ float red[NB];
    int t = threadIdx.x;
    if (t < NB) {
        float cnt = scnt[t];
        float lp[NCLS];
        float mx = -1e30f;
        #pragma unroll
        for (int c = 0; c < NCLS; ++c) {
            lp[c] = ssum[t * NCLS + c] / cnt;
            out[1 + t * NCLS + c] = lp[c];
            mx = fmaxf(mx, lp[c]);
        }
        float se = 0.0f;
        #pragma unroll
        for (int c = 0; c < NCLS; ++c) se += __expf(lp[c] - mx);
        float lse = mx + __logf(se);
        int yy = y[t];
        float lpy = 0.0f;
        #pragma unroll
        for (int c = 0; c < NCLS; ++c) if (c == yy) lpy = lp[c];
        red[t] = -(lpy - lse);
    }
    __syncthreads();
    if (t == 0) {
        float s = 0.0f;
        for (int b = 0; b < NB; ++b) s += red[b];
        out[0] = s / (float)NB;
    }
}

__global__ void zero_ws(float* __restrict__ p, int n) {
    int i = blockIdx.x * blockDim.x + threadIdx.x;
    if (i < n) p[i] = 0.0f;
}

__global__ void cvt_w(const float* __restrict__ s, __bf16* __restrict__ d, int n) {
    int i = blockIdx.x * blockDim.x + threadIdx.x;
    if (i < n) d[i] = (__bf16)s[i];
}

// ---------------------------------------------------------------------------
extern "C" void kernel_launch(void* const* d_in, const int* in_sizes, int n_in,
                              void* d_out, int out_size, void* d_ws, size_t ws_size,
                              hipStream_t stream)
{
    const float* x        = (const float*)d_in[0];
    const int*   y        = (const int*)  d_in[1];
    const int*   segids   = (const int*)  d_in[2];
    const float* enc_i_w  = (const float*)d_in[3];
    const float* enc_i_b  = (const float*)d_in[4];
    const float* enc_h0_w = (const float*)d_in[5];
    const float* enc_h0_b = (const float*)d_in[6];
    const float* enc_z_w  = (const float*)d_in[7];
    const float* enc_z_b  = (const float*)d_in[8];
    const float* imp_i_w  = (const float*)d_in[9];
    const float* imp_i_b  = (const float*)d_in[10];
    const float* imp_h0_w = (const float*)d_in[11];
    const float* imp_h0_b = (const float*)d_in[12];
    const float* imp_p_w  = (const float*)d_in[13];
    const float* imp_p_b  = (const float*)d_in[14];
    const float* proto    = (const float*)d_in[15];

    // workspace layout
    __bf16* Ybf     = (__bf16*)d_ws;                         // [50000,256] bf16
    __bf16* w_eh_bf = Ybf + (size_t)NCELLS * NT;             // 128*128
    __bf16* w_ez_bf = w_eh_bf + HDIM * HDIM;                 // 64*128
    __bf16* w_ih_bf = w_ez_bf + ZDIM * HDIM;                 // 128*128
    __bf16* w_ip_bf = w_ih_bf + HDIM * HDIM;                 // 256*128
    float*  ssum    = (float*)(w_ip_bf + 2 * HDIM * HDIM);   // [64,8]
    float*  scnt    = ssum + NB * NCLS;                      // [64]

    zero_ws<<<3, 256, 0, stream>>>(ssum, NB * NCLS + NB);

    cvt_w<<<(HDIM * HDIM + 255) / 256, 256, 0, stream>>>(enc_h0_w, w_eh_bf, HDIM * HDIM);
    cvt_w<<<(ZDIM * HDIM + 255) / 256, 256, 0, stream>>>(enc_z_w,  w_ez_bf, ZDIM * HDIM);
    cvt_w<<<(HDIM * HDIM + 255) / 256, 256, 0, stream>>>(imp_h0_w, w_ih_bf, HDIM * HDIM);
    cvt_w<<<(2 * HDIM * HDIM + 255) / 256, 256, 0, stream>>>(imp_p_w, w_ip_bf, 2 * HDIM * HDIM);

    const int mblocks = (NCELLS + MT - 1) / MT;       // 391
    fused_layer1<<<mblocks, 512, 0, stream>>>(x, enc_i_w, enc_i_b, imp_i_w, imp_i_b, Ybf);

    const int ngroups = NCELLS / 16;                  // 3125
    const int tblocks = (ngroups + WPB - 1) / WPB;    // 782
    fused_tail<<<tblocks, 128, 0, stream>>>(Ybf, segids,
                                            w_eh_bf, enc_h0_b, w_ez_bf, enc_z_b,
                                            w_ih_bf, imp_h0_b, w_ip_bf, imp_p_b,
                                            proto, ssum, scnt, ngroups);

    finalize_kernel<<<1, 64, 0, stream>>>(ssum, scnt, y, (float*)d_out);
}
